// TipAdapterFLinear_67422396612890
// MI455X (gfx1250) — compile-verified, matrix operand order checked by reading, not run on hardware
//
#include <hip/hip_runtime.h>
#include <hip/hip_bf16.h>

typedef __attribute__((ext_vector_type(16))) __bf16 v16bf;
typedef __attribute__((ext_vector_type(8)))  float  v8f;

#define B_ROWS 8192
#define D_IN   1024
#define D_OUT  768
#define N_KEYS 16384
#define PN_STRIDE 776   // 768 + 8 bf16 pad: row stride = 388 dwords => +4 banks/row

// ---- Fragment loaders (CDNA5 wave32 16-bit A/B layout) ----------------------
// Per ISA 7.12.2: lane m (0-15) holds K = kh..kh+7 and kh+16..kh+23 where
// kh = (lane>>4)*8. So a frag = 8 contiguous bf16 at p and 8 at p+16.
static __device__ __forceinline__ v16bf load_frag_bf16(const __bf16* p) {
  union { v16bf v; uint4 q[2]; } u;
  u.q[0] = *reinterpret_cast<const uint4*>(p);
  u.q[1] = *reinterpret_cast<const uint4*>(p + 16);
  return u.v;
}

static __device__ __forceinline__ v16bf load_frag_f32(const float* p) {
  union { v16bf v; __bf16 e[16]; } u;
  float4 a0 = *reinterpret_cast<const float4*>(p);
  float4 a1 = *reinterpret_cast<const float4*>(p + 4);
  float4 b0 = *reinterpret_cast<const float4*>(p + 16);
  float4 b1 = *reinterpret_cast<const float4*>(p + 20);
  u.e[0]  = (__bf16)a0.x; u.e[1]  = (__bf16)a0.y; u.e[2]  = (__bf16)a0.z; u.e[3]  = (__bf16)a0.w;
  u.e[4]  = (__bf16)a1.x; u.e[5]  = (__bf16)a1.y; u.e[6]  = (__bf16)a1.z; u.e[7]  = (__bf16)a1.w;
  u.e[8]  = (__bf16)b0.x; u.e[9]  = (__bf16)b0.y; u.e[10] = (__bf16)b0.z; u.e[11] = (__bf16)b0.w;
  u.e[12] = (__bf16)b1.x; u.e[13] = (__bf16)b1.y; u.e[14] = (__bf16)b1.z; u.e[15] = (__bf16)b1.w;
  return u.v;
}

// Cooperative copy of one 16x768 bf16 query tile into padded LDS.
static __device__ __forceinline__ void stage_pn_tile(__bf16* pnS, const __bf16* pn,
                                                     int rowBase, int tid) {
  for (int i = tid; i < 16 * (D_OUT / 8); i += 256) {     // 96 uint4 per row
    const int row = i / (D_OUT / 8);
    const int c8  = (i - row * (D_OUT / 8)) * 8;
    *reinterpret_cast<uint4*>(&pnS[row * PN_STRIDE + c8]) =
        *reinterpret_cast<const uint4*>(&pn[(size_t)(rowBase + row) * D_OUT + c8]);
  }
}

// ---- Kernel 0a: keys fp32 -> bf16 (row-major, L2-resident 24MB) -------------
__global__ void cvt_keys_kernel(const float* __restrict__ src, __bf16* __restrict__ dst, int count) {
  int i = blockIdx.x * blockDim.x + threadIdx.x;
  if (i < count) dst[i] = (__bf16)src[i];
}

// ---- Kernel 0b: values fp32 -> bf16 TRANSPOSED [D_OUT][N] -------------------
__global__ void transpose_vals_kernel(const float* __restrict__ vals, __bf16* __restrict__ valsT) {
  int i = blockIdx.x * blockDim.x + threadIdx.x;          // over D_OUT*N
  if (i < D_OUT * N_KEYS) {
    int o = i / N_KEYS;
    int n = i - o * N_KEYS;
    valsT[i] = (__bf16)vals[(size_t)n * D_OUT + o];
  }
}

// ---- Kernel 1: projected = x @ W^T + b (bf16 WMMA, fp32 accum) --------------
__global__ void __launch_bounds__(256) proj_gemm_kernel(const float* __restrict__ x,
                                                        const float* __restrict__ W,
                                                        const float* __restrict__ bias,
                                                        float* __restrict__ out) {
  const int wave = threadIdx.x >> 5;
  const int lane = threadIdx.x & 31;
  const int r    = lane & 15;
  const int kh   = (lane >> 4) * 8;
  const int rowBase = blockIdx.x * 128 + wave * 16;
  const int colBase = blockIdx.y * 64;

  v8f acc[4] = {};
  const float* xr = x + (size_t)(rowBase + r) * D_IN + kh;
  for (int k0 = 0; k0 < D_IN; k0 += 32) {
    v16bf a = load_frag_f32(xr + k0);
#pragma unroll
    for (int t = 0; t < 4; ++t) {
      v16bf b = load_frag_f32(W + (size_t)(colBase + t * 16 + r) * D_IN + k0 + kh);
      acc[t] = __builtin_amdgcn_wmma_f32_16x16x32_bf16(false, a, false, b, (short)0, acc[t], false, false);
    }
  }
#pragma unroll
  for (int t = 0; t < 4; ++t) {
    const int col = colBase + t * 16 + r;
    const float bv = bias[col];
#pragma unroll
    for (int j = 0; j < 8; ++j)
      out[(size_t)(rowBase + j + kh) * D_OUT + col] = acc[t][j] + bv;
  }
}

// ---- Kernel 2: L2-normalize rows of projected -> pn (bf16) ------------------
__global__ void __launch_bounds__(256) l2norm_kernel(const float* __restrict__ proj,
                                                     __bf16* __restrict__ pn) {
  __shared__ float red[256];
  const int row = blockIdx.x;
  const float* p = proj + (size_t)row * D_OUT;
  float s = 0.f;
  for (int c = threadIdx.x; c < D_OUT; c += 256) { float v = p[c]; s += v * v; }
  red[threadIdx.x] = s;
  __syncthreads();
  for (int off = 128; off > 0; off >>= 1) {
    if (threadIdx.x < off) red[threadIdx.x] += red[threadIdx.x + off];
    __syncthreads();
  }
  const float inv = 1.0f / fmaxf(sqrtf(red[0]), 1e-12f);
  for (int c = threadIdx.x; c < D_OUT; c += 256)
    pn[(size_t)row * D_OUT + c] = (__bf16)(p[c] * inv);
}

// ---- Kernel 3: softmax pass 1 — per-row running (max, sumexp) ---------------
// Block = 16 rows staged in LDS; 8 waves partition the key dim in 16-key
// slices of each 128-key chunk. Cross-wave combine via LDS at the end.
__global__ void __launch_bounds__(256) softmax_stats_kernel(const __bf16* __restrict__ pn,
                                                            const __bf16* __restrict__ keys,
                                                            const float* __restrict__ betaPtr,
                                                            float* __restrict__ mrow,
                                                            float* __restrict__ lrow) {
  __shared__ __align__(16) __bf16 pnS[16 * PN_STRIDE];
  __shared__ float sm[16][8];
  __shared__ float sl[16][8];
  const int wave = threadIdx.x >> 5, lane = threadIdx.x & 31;
  const int r = lane & 15, kh = (lane >> 4) * 8;
  const int rowBase = blockIdx.x * 16;
  const float beta = betaPtr[0];

  stage_pn_tile(pnS, pn, rowBase, threadIdx.x);
  __syncthreads();

  float m8[8], l8[8];
#pragma unroll
  for (int j = 0; j < 8; ++j) { m8[j] = -1e30f; l8[j] = 0.f; }

  const __bf16* aBase = &pnS[r * PN_STRIDE + kh];
  for (int n0 = 0; n0 < N_KEYS; n0 += 128) {
    const __bf16* bBase = keys + (size_t)(n0 + wave * 16 + r) * D_OUT + kh;
    if (n0 + 128 < N_KEYS)                              // warm L0 for next chunk
      __builtin_prefetch(bBase + 128 * D_OUT, 0, 1);
    v8f c = {};
    for (int k0 = 0; k0 < D_OUT; k0 += 32) {
      v16bf a = load_frag_bf16(aBase + k0);             // ds_load_b128 x2
      v16bf b = load_frag_bf16(bBase + k0);             // global_load_b128 x2
      c = __builtin_amdgcn_wmma_f32_16x16x32_bf16(false, a, false, b, (short)0, c, false, false);
    }
#pragma unroll
    for (int j = 0; j < 8; ++j) {
      float s = beta * c[j];
      float mx = s;                                     // row lives across a 16-lane group
      mx = fmaxf(mx, __shfl_xor(mx, 1, 32));
      mx = fmaxf(mx, __shfl_xor(mx, 2, 32));
      mx = fmaxf(mx, __shfl_xor(mx, 4, 32));
      mx = fmaxf(mx, __shfl_xor(mx, 8, 32));
      float mnew = fmaxf(m8[j], mx);
      float e = __expf(s - mnew);
      e += __shfl_xor(e, 1, 32);
      e += __shfl_xor(e, 2, 32);
      e += __shfl_xor(e, 4, 32);
      e += __shfl_xor(e, 8, 32);
      l8[j] = l8[j] * __expf(m8[j] - mnew) + e;
      m8[j] = mnew;
    }
  }
  if ((lane & 15) == 0) {
#pragma unroll
    for (int j = 0; j < 8; ++j) { sm[j + kh][wave] = m8[j]; sl[j + kh][wave] = l8[j]; }
  }
  __syncthreads();
  if (threadIdx.x < 16) {
    const int row = threadIdx.x;
    float M = -1e30f;
    for (int w = 0; w < 8; ++w) M = fmaxf(M, sm[row][w]);
    float L = 0.f;
    for (int w = 0; w < 8; ++w) L += sl[row][w] * __expf(sm[row][w] - M);
    mrow[rowBase + row] = M;
    lrow[rowBase + row] = L;
  }
}

// ---- Kernel 4: pass 2 — recompute S, P=exp(βS−m), O += P@V; out += αO/l -----
__global__ void __launch_bounds__(256) retrieve_kernel(const __bf16* __restrict__ pn,
                                                       const __bf16* __restrict__ keys,
                                                       const __bf16* __restrict__ valsT,
                                                       const float* __restrict__ mrow,
                                                       const float* __restrict__ lrow,
                                                       const float* __restrict__ betaPtr,
                                                       const float* __restrict__ alphaPtr,
                                                       float* __restrict__ out) {
  __shared__ __align__(16) __bf16 pnS[16 * PN_STRIDE];
  __shared__ __align__(16) __bf16 P[16 * 136];          // stride 136 => conflict-free b128
  const int wave = threadIdx.x >> 5, lane = threadIdx.x & 31;
  const int r = lane & 15, kh = (lane >> 4) * 8;
  const int rowBase = blockIdx.x * 16;
  const float beta = betaPtr[0], alpha = alphaPtr[0];

  float m8[8], linv[8];
#pragma unroll
  for (int j = 0; j < 8; ++j) {
    m8[j]   = mrow[rowBase + j + kh];
    linv[j] = alpha / lrow[rowBase + j + kh];
  }

  stage_pn_tile(pnS, pn, rowBase, threadIdx.x);
  __syncthreads();

  v8f o[6] = {};
  const __bf16* aBase = &pnS[r * PN_STRIDE + kh];
  for (int n0 = 0; n0 < N_KEYS; n0 += 128) {
    // S tile for this wave's 16 keys
    const __bf16* bBase = keys + (size_t)(n0 + wave * 16 + r) * D_OUT + kh;
    if (n0 + 128 < N_KEYS)
      __builtin_prefetch(bBase + 128 * D_OUT, 0, 1);
    v8f c = {};
    for (int k0 = 0; k0 < D_OUT; k0 += 32) {
      v16bf a = load_frag_bf16(aBase + k0);
      v16bf b = load_frag_bf16(bBase + k0);
      c = __builtin_amdgcn_wmma_f32_16x16x32_bf16(false, a, false, b, (short)0, c, false, false);
    }
#pragma unroll
    for (int j = 0; j < 8; ++j) {
      float p = __expf(beta * c[j] - m8[j]);
      P[(j + kh) * 136 + wave * 16 + r] = (__bf16)p;    // C layout -> A layout in LDS
    }
    __syncthreads();
    // O[16 x 96-slice] += P(16x128) @ V(128 x 96-slice)
#pragma unroll
    for (int kk = 0; kk < 4; ++kk) {
      v16bf a = load_frag_bf16(&P[r * 136 + kk * 32 + kh]);
#pragma unroll
      for (int t = 0; t < 6; ++t) {
        const int col = wave * 96 + t * 16 + r;
        v16bf b = load_frag_bf16(valsT + (size_t)col * N_KEYS + n0 + kk * 32 + kh);
        o[t] = __builtin_amdgcn_wmma_f32_16x16x32_bf16(false, a, false, b, (short)0, o[t], false, false);
      }
    }
    __syncthreads();                                    // P is overwritten next iteration
  }
#pragma unroll
  for (int t = 0; t < 6; ++t) {
    const int col = wave * 96 + t * 16 + r;
#pragma unroll
    for (int j = 0; j < 8; ++j) {
      const size_t idx = (size_t)(rowBase + j + kh) * D_OUT + col;
      out[idx] += o[t][j] * linv[j];                    // out already holds `projected`
    }
  }
}

// ---- Host launch ------------------------------------------------------------
extern "C" void kernel_launch(void* const* d_in, const int* in_sizes, int n_in,
                              void* d_out, int out_size, void* d_ws, size_t ws_size,
                              hipStream_t stream) {
  const float* x     = (const float*)d_in[0];
  const float* W     = (const float*)d_in[1];
  const float* bias  = (const float*)d_in[2];
  const float* keys  = (const float*)d_in[3];
  const float* vals  = (const float*)d_in[4];
  const float* beta  = (const float*)d_in[5];
  const float* alpha = (const float*)d_in[6];
  float* out = (float*)d_out;

  // Workspace layout (~60 MB): keys bf16 | valuesT bf16 | pn bf16 | m | l
  __bf16* keysB = (__bf16*)d_ws;
  __bf16* valsT = keysB + (size_t)N_KEYS * D_OUT;
  __bf16* pnB   = valsT + (size_t)D_OUT * N_KEYS;
  float*  mrow  = (float*)(pnB + (size_t)B_ROWS * D_OUT);
  float*  lrow  = mrow + B_ROWS;

  const int cnt = N_KEYS * D_OUT;
  cvt_keys_kernel<<<(cnt + 255) / 256, 256, 0, stream>>>(keys, keysB, cnt);
  transpose_vals_kernel<<<(cnt + 255) / 256, 256, 0, stream>>>(vals, valsT);

  proj_gemm_kernel<<<dim3(B_ROWS / 128, D_OUT / 64), 256, 0, stream>>>(x, W, bias, out);
  l2norm_kernel<<<B_ROWS, 256, 0, stream>>>(out, pnB);
  softmax_stats_kernel<<<B_ROWS / 16, 256, 0, stream>>>(pnB, keysB, beta, mrow, lrow);
  retrieve_kernel<<<B_ROWS / 16, 256, 0, stream>>>(pnB, keysB, valsT, mrow, lrow, beta, alpha, out);
}
